// VQVAE_10050223473100
// MI455X (gfx1250) — compile-verified, matrix operand order checked by reading, not run on hardware
//
#include <hip/hip_runtime.h>
#include <hip/hip_bf16.h>

// ---------------------------------------------------------------------------
// Types for WMMA
// ---------------------------------------------------------------------------
typedef __attribute__((ext_vector_type(16))) __bf16 v16bf;
typedef __attribute__((ext_vector_type(8)))  float  v8f;
typedef __attribute__((ext_vector_type(4)))  int    v4i;

struct alignas(16) B128 { unsigned int w[4]; };
union FragU { B128 q[2]; v16bf v; };

#define BM 128
#define BN 128
#define BK 32
#define LDT 40   // padded LDS row stride in bf16 elems (80 B = 16B aligned, bank-spread)

// Model dims (fixed by the reference)
#define BATCH      16384
#define INPUT_DIM  1024
#define HIDDEN_DIM 2048
#define LATENT_DIM 256
#define NUM_EMB    1024

// ---------------------------------------------------------------------------
// CDNA5 async global->LDS copy (ASYNCcnt path), with compile-safe fallbacks
// ---------------------------------------------------------------------------
#if __has_builtin(__builtin_amdgcn_global_load_async_to_lds_b128)
#define HAVE_ASYNC_LDS 1
#else
#define HAVE_ASYNC_LDS 0
#endif

#if __has_builtin(__builtin_amdgcn_s_wait_asynccnt)
#define WAIT_ASYNC(n) __builtin_amdgcn_s_wait_asynccnt(n)
#else
#define WAIT_ASYNC(n) asm volatile("s_wait_asynccnt %0" ::"n"(n))
#endif

#if HAVE_ASYNC_LDS
typedef __attribute__((address_space(1))) v4i* gv4i_p;  // global v4i*
typedef __attribute__((address_space(3))) v4i* lv4i_p;  // LDS v4i*

__device__ __forceinline__ void async_copy_b128(const __hip_bfloat16* g,
                                                __hip_bfloat16* l) {
  __builtin_amdgcn_global_load_async_to_lds_b128((gv4i_p)g, (lv4i_p)l, 0, 0);
}
#endif

// ---------------------------------------------------------------------------
// f32 -> bf16 conversion
// ---------------------------------------------------------------------------
__global__ __launch_bounds__(256) void k_f32_to_bf16(const float* __restrict__ in,
                                                     __hip_bfloat16* __restrict__ out,
                                                     int n) {
  int i = blockIdx.x * 256 + threadIdx.x;
  if (i < n) out[i] = __float2bfloat16(in[i]);
}

// emb [K=1024, D=256] -> embT [D=256, K=1024]
__global__ __launch_bounds__(256) void k_transpose_emb(const float* __restrict__ emb,
                                                       float* __restrict__ embT) {
  int i = blockIdx.x * 256 + threadIdx.x;     // 0 .. 262143
  int k = i >> 8;
  int d = i & 255;
  embT[d * NUM_EMB + k] = emb[i];
}

// ---------------------------------------------------------------------------
// Tiled bf16 WMMA GEMM:  C[M,N] = act(A[M,K] @ W[N,K]^T + bias[N])
// Block tile 128x128x32, 256 threads (8 waves); wave -> 64x32 sub-tile
// (4 M-frags x 2 N-frags of 16x16), 8 WMMA per K-step.
// Double-buffered LDS filled by async global->LDS b128 copies (ASYNCcnt),
// so tile k+1 streams from HBM/L2 while tile k runs on the WMMA pipes.
// ---------------------------------------------------------------------------
__device__ __forceinline__ v16bf lds_frag(const __hip_bfloat16* tile, int row, int half) {
  // ISA 7.12.2 bf16 16x32 layout: lanes 0-15 take K[0..7],[16..23],
  // lanes 16-31 take K[8..15],[24..31] (half = lane>>4).
  const __hip_bfloat16* r = tile + row * LDT;
  FragU u;
  u.q[0] = *(const B128*)(r + half * 8);
  u.q[1] = *(const B128*)(r + 16 + half * 8);
  return u.v;
}

__device__ __forceinline__ void store_out(float* p, float v) { *p = v; }
__device__ __forceinline__ void store_out(__hip_bfloat16* p, float v) {
  *p = __float2bfloat16(v);
}

template <typename OutT, bool RELU>
__global__ __launch_bounds__(256)
void k_gemm_bf16_wmma(const __hip_bfloat16* __restrict__ A,
                      const __hip_bfloat16* __restrict__ W,
                      const float* __restrict__ bias,
                      OutT* __restrict__ C,
                      int M, int N, int K) {
  __shared__ __hip_bfloat16 sA[2][BM * LDT];
  __shared__ __hip_bfloat16 sW[2][BN * LDT];

  const int tid  = threadIdx.x;
  const int lane = tid & 31;
  const int wave = tid >> 5;        // 0..7
  const int wm   = wave >> 2;       // 0..1  -> 64-row band
  const int wn   = wave & 3;        // 0..3  -> 32-col band

  const int m0 = blockIdx.y * BM;
  const int n0 = blockIdx.x * BN;

  // cooperative tile loading: thread -> one row-half (16 bf16 = 32 B = 2x b128)
  const int ldRow = tid >> 1;            // 0..127
  const int ldOff = (tid & 1) * 16;      // elem 0 or 16 within K-tile

  const __hip_bfloat16* gA = A + (size_t)(m0 + ldRow) * K + ldOff;
  const __hip_bfloat16* gW = W + (size_t)(n0 + ldRow) * K + ldOff;
  __hip_bfloat16* lA = &sA[0][ldRow * LDT + ldOff];
  __hip_bfloat16* lW = &sW[0][ldRow * LDT + ldOff];
  const int lstride = BM * LDT;          // elems between buffer 0 and 1

  v8f acc[4][2];
  const v8f vzero = {0.f, 0.f, 0.f, 0.f, 0.f, 0.f, 0.f, 0.f};
#pragma unroll
  for (int i = 0; i < 4; ++i)
#pragma unroll
    for (int j = 0; j < 2; ++j) acc[i][j] = vzero;

  const int ksteps = K / BK;
  const int half = lane >> 4;
  const int l16  = lane & 15;

#if HAVE_ASYNC_LDS
  // ---- async double-buffered pipeline ----
  {
    // prologue: stage tile 0 into buffer 0 (4 async b128 per thread)
    async_copy_b128(gA, lA);
    async_copy_b128(gA + 8, lA + 8);
    async_copy_b128(gW, lW);
    async_copy_b128(gW + 8, lW + 8);
  }
  for (int kt = 0; kt < ksteps; ++kt) {
    const int cur = kt & 1;
    if (kt + 1 < ksteps) {
      const int nxt = cur ^ 1;
      const __hip_bfloat16* a = gA + (size_t)(kt + 1) * BK;
      const __hip_bfloat16* w = gW + (size_t)(kt + 1) * BK;
      async_copy_b128(a, lA + nxt * lstride);
      async_copy_b128(a + 8, lA + nxt * lstride + 8);
      async_copy_b128(w, lW + nxt * lstride);
      async_copy_b128(w + 8, lW + nxt * lstride + 8);
      WAIT_ASYNC(4);   // async loads retire in order: current tile resident
    } else {
      WAIT_ASYNC(0);
    }
    __syncthreads();

    const __hip_bfloat16* tA = sA[cur];
    const __hip_bfloat16* tW = sW[cur];
    v16bf bfrag[2];
#pragma unroll
    for (int fn = 0; fn < 2; ++fn)
      bfrag[fn] = lds_frag(tW, wn * 32 + fn * 16 + l16, half);
#pragma unroll
    for (int fm = 0; fm < 4; ++fm) {
      v16bf afrag = lds_frag(tA, wm * 64 + fm * 16 + l16, half);
#pragma unroll
      for (int fn = 0; fn < 2; ++fn) {
        acc[fm][fn] = __builtin_amdgcn_wmma_f32_16x16x32_bf16(
            false, afrag, false, bfrag[fn], (short)0, acc[fm][fn], false, false);
      }
    }
    __syncthreads();   // buffer free before it is refilled next iteration
  }
#else
  // ---- fallback: register-staged single buffer ----
  for (int kt = 0; kt < ksteps; ++kt) {
    __syncthreads();
    B128 a0 = *(const B128*)(gA);
    B128 a1 = *(const B128*)(gA + 8);
    B128 w0 = *(const B128*)(gW);
    B128 w1 = *(const B128*)(gW + 8);
    if (kt + 1 < ksteps) {
      __builtin_prefetch(gA + BK, 0, 1);
      __builtin_prefetch(gW + BK, 0, 1);
    }
    *(B128*)(lA) = a0;
    *(B128*)(lA + 8) = a1;
    *(B128*)(lW) = w0;
    *(B128*)(lW + 8) = w1;
    gA += BK;
    gW += BK;
    __syncthreads();

    v16bf bfrag[2];
#pragma unroll
    for (int fn = 0; fn < 2; ++fn)
      bfrag[fn] = lds_frag(sW[0], wn * 32 + fn * 16 + l16, half);
#pragma unroll
    for (int fm = 0; fm < 4; ++fm) {
      v16bf afrag = lds_frag(sA[0], wm * 64 + fm * 16 + l16, half);
#pragma unroll
      for (int fn = 0; fn < 2; ++fn) {
        acc[fm][fn] = __builtin_amdgcn_wmma_f32_16x16x32_bf16(
            false, afrag, false, bfrag[fn], (short)0, acc[fm][fn], false, false);
      }
    }
  }
#endif

  // Epilogue: C/D layout (ISA 7.12.2): VGPR r -> M = r (+8 for lanes>=16), N = lane%16
  const int mHi = half * 8;
#pragma unroll
  for (int fn = 0; fn < 2; ++fn) {
    const int gn = n0 + wn * 32 + fn * 16 + l16;
    const float bv = bias[gn];
#pragma unroll
    for (int fm = 0; fm < 4; ++fm) {
      const int gmBase = m0 + wm * 64 + fm * 16 + mHi;
#pragma unroll
      for (int r = 0; r < 8; ++r) {
        float v = acc[fm][fn][r] + bv;
        if (RELU) v = fmaxf(v, 0.0f);
        store_out(C + (size_t)(gmBase + r) * (size_t)N + gn, v);
      }
    }
  }
}

// ---------------------------------------------------------------------------
// VQ: per-row argmin over 1024 codes (256-d), lowest-index tie-break.
// 8 rows per workgroup (codebook stays L2-resident; reads via transposed emb
// are coalesced). Writes z_q (f32 straight-through forward value + bf16 for
// the decoder) and per-row min squared distance.
// ---------------------------------------------------------------------------
#define VQ_ROWS 8
__global__ __launch_bounds__(256)
void k_vq(const float* __restrict__ Z, const float* __restrict__ emb,
          const float* __restrict__ embT, float* __restrict__ zq_f32,
          __hip_bfloat16* __restrict__ zq_bf16, float* __restrict__ rowmin) {
  __shared__ float zrow[VQ_ROWS][LATENT_DIM];
  __shared__ float sdist[VQ_ROWS][NUM_EMB];   // 32 KB
  __shared__ float rv[256];
  __shared__ int   rk[256];
  __shared__ int   skbest[VQ_ROWS];

  const int t    = threadIdx.x;
  const int row0 = blockIdx.x * VQ_ROWS;

#pragma unroll
  for (int i = 0; i < VQ_ROWS; ++i)
    zrow[i][t] = Z[(size_t)(row0 + i) * LATENT_DIM + t];
  __syncthreads();

  for (int c = 0; c < 4; ++c) {
    const int k = c * 256 + t;
    float acc[VQ_ROWS];
#pragma unroll
    for (int i = 0; i < VQ_ROWS; ++i) acc[i] = 0.0f;
    for (int d = 0; d < LATENT_DIM; ++d) {
      const float e = embT[d * NUM_EMB + k];
#pragma unroll
      for (int i = 0; i < VQ_ROWS; ++i) {
        float df = zrow[i][d] - e;
        acc[i] = fmaf(df, df, acc[i]);
      }
    }
#pragma unroll
    for (int i = 0; i < VQ_ROWS; ++i) sdist[i][k] = acc[i];
  }
  __syncthreads();

  for (int i = 0; i < VQ_ROWS; ++i) {
    float best = 3.402823e38f;
    int bk = NUM_EMB;
    for (int c = 0; c < 4; ++c) {
      int k = c * 256 + t;
      float v = sdist[i][k];
      if (v < best || (v == best && k < bk)) { best = v; bk = k; }
    }
    rv[t] = best;
    rk[t] = bk;
    __syncthreads();
    for (int s = 128; s > 0; s >>= 1) {
      if (t < s) {
        float v2 = rv[t + s];
        int k2 = rk[t + s];
        if (v2 < rv[t] || (v2 == rv[t] && k2 < rk[t])) { rv[t] = v2; rk[t] = k2; }
      }
      __syncthreads();
    }
    if (t == 0) {
      skbest[i] = rk[0];
      rowmin[row0 + i] = rv[0];
    }
    __syncthreads();
  }

#pragma unroll
  for (int i = 0; i < VQ_ROWS; ++i) {
    int bk = skbest[i];
    float e = emb[(size_t)bk * LATENT_DIM + t];
    size_t off = (size_t)(row0 + i) * LATENT_DIM + t;
    zq_f32[off]  = e;
    zq_bf16[off] = __float2bfloat16(e);
  }
}

// ---------------------------------------------------------------------------
// Deterministic reductions for the losses
// ---------------------------------------------------------------------------
__global__ __launch_bounds__(256)
void k_sqdiff_partial(const float* __restrict__ a, const float* __restrict__ b,
                      float* __restrict__ partial, int n) {
  __shared__ float s[256];
  const int t = threadIdx.x;
  float acc = 0.0f;
  for (int i = blockIdx.x * 256 + t; i < n; i += gridDim.x * 256) {
    float d = a[i] - b[i];
    acc = fmaf(d, d, acc);
  }
  s[t] = acc;
  __syncthreads();
  for (int sh = 128; sh > 0; sh >>= 1) {
    if (t < sh) s[t] += s[t + sh];
    __syncthreads();
  }
  if (t == 0) partial[blockIdx.x] = s[0];
}

__global__ __launch_bounds__(256)
void k_finalize(const float* __restrict__ rowmin, int nrow,
                const float* __restrict__ partial, int npart,
                float* __restrict__ out_total, float* __restrict__ out_vq,
                float inv_recon_den, float vq_scale) {
  __shared__ float s[256];
  const int t = threadIdx.x;
  float a = 0.0f, b = 0.0f;
  for (int i = t; i < nrow; i += 256) a += rowmin[i];
  for (int i = t; i < npart; i += 256) b += partial[i];

  s[t] = a;
  __syncthreads();
  for (int sh = 128; sh > 0; sh >>= 1) { if (t < sh) s[t] += s[t + sh]; __syncthreads(); }
  float vqsum = s[0];
  __syncthreads();

  s[t] = b;
  __syncthreads();
  for (int sh = 128; sh > 0; sh >>= 1) { if (t < sh) s[t] += s[t + sh]; __syncthreads(); }
  float rsum = s[0];

  if (t == 0) {
    float vq = vq_scale * vqsum;              // 1.25 * mean((z_q - z)^2)
    float recon = rsum * inv_recon_den;       // mean((x_recon - x)^2)
    *out_vq = vq;
    *out_total = recon + vq;
  }
}

// ---------------------------------------------------------------------------
// Host-side orchestration
// ---------------------------------------------------------------------------
static inline size_t align256(size_t v) { return (v + 255) & ~(size_t)255; }

extern "C" void kernel_launch(void* const* d_in, const int* in_sizes, int n_in,
                              void* d_out, int out_size, void* d_ws, size_t ws_size,
                              hipStream_t stream) {
  (void)in_sizes; (void)n_in; (void)out_size; (void)ws_size;

  const float* x   = (const float*)d_in[0];
  const float* ew1 = (const float*)d_in[1];  const float* eb1 = (const float*)d_in[2];
  const float* ew2 = (const float*)d_in[3];  const float* eb2 = (const float*)d_in[4];
  const float* ew3 = (const float*)d_in[5];  const float* eb3 = (const float*)d_in[6];
  const float* dw1 = (const float*)d_in[7];  const float* db1 = (const float*)d_in[8];
  const float* dw2 = (const float*)d_in[9];  const float* db2 = (const float*)d_in[10];
  const float* dw3 = (const float*)d_in[11]; const float* db3 = (const float*)d_in[12];
  const float* emb = (const float*)d_in[13];

  float* out      = (float*)d_out;
  float* x_recon  = out;                                              // [16384*1024]
  float* total_l  = out + (size_t)BATCH * INPUT_DIM;                  // [1]
  float* vq_l     = total_l + 1;                                      // [1]
  float* zq_out   = vq_l + 1;                                         // [16384*256]

  char* p = (char*)d_ws;
  auto alloc = [&](size_t bytes) { char* r = p; p += align256(bytes); return r; };

  __hip_bfloat16* xb  = (__hip_bfloat16*)alloc((size_t)BATCH * INPUT_DIM * 2);
  __hip_bfloat16* w1b = (__hip_bfloat16*)alloc((size_t)HIDDEN_DIM * INPUT_DIM * 2);
  __hip_bfloat16* w2b = (__hip_bfloat16*)alloc((size_t)HIDDEN_DIM * HIDDEN_DIM * 2);
  __hip_bfloat16* w3b = (__hip_bfloat16*)alloc((size_t)LATENT_DIM * HIDDEN_DIM * 2);
  __hip_bfloat16* d1b = (__hip_bfloat16*)alloc((size_t)HIDDEN_DIM * LATENT_DIM * 2);
  __hip_bfloat16* d2b = (__hip_bfloat16*)alloc((size_t)HIDDEN_DIM * HIDDEN_DIM * 2);
  __hip_bfloat16* d3b = (__hip_bfloat16*)alloc((size_t)INPUT_DIM * HIDDEN_DIM * 2);
  __hip_bfloat16* h1  = (__hip_bfloat16*)alloc((size_t)BATCH * HIDDEN_DIM * 2);
  __hip_bfloat16* h2  = (__hip_bfloat16*)alloc((size_t)BATCH * HIDDEN_DIM * 2);
  float*          zf  = (float*)alloc((size_t)BATCH * LATENT_DIM * 4);
  __hip_bfloat16* zqb = (__hip_bfloat16*)alloc((size_t)BATCH * LATENT_DIM * 2);
  float*          embT = (float*)alloc((size_t)NUM_EMB * LATENT_DIM * 4);
  float*          rowmin = (float*)alloc((size_t)BATCH * 4);
  const int NPART = 4096;
  float*          part = (float*)alloc((size_t)NPART * 4);

  auto cvt = [&](const float* src, __hip_bfloat16* dst, size_t n) {
    k_f32_to_bf16<<<(unsigned)((n + 255) / 256), 256, 0, stream>>>(src, dst, (int)n);
  };
  cvt(x,   xb,  (size_t)BATCH * INPUT_DIM);
  cvt(ew1, w1b, (size_t)HIDDEN_DIM * INPUT_DIM);
  cvt(ew2, w2b, (size_t)HIDDEN_DIM * HIDDEN_DIM);
  cvt(ew3, w3b, (size_t)LATENT_DIM * HIDDEN_DIM);
  cvt(dw1, d1b, (size_t)HIDDEN_DIM * LATENT_DIM);
  cvt(dw2, d2b, (size_t)HIDDEN_DIM * HIDDEN_DIM);
  cvt(dw3, d3b, (size_t)INPUT_DIM * HIDDEN_DIM);
  k_transpose_emb<<<(NUM_EMB * LATENT_DIM) / 256, 256, 0, stream>>>(emb, embT);

  // Encoder
  {
    dim3 g1(HIDDEN_DIM / BN, BATCH / BM);
    k_gemm_bf16_wmma<__hip_bfloat16, true><<<g1, 256, 0, stream>>>(
        xb, w1b, eb1, h1, BATCH, HIDDEN_DIM, INPUT_DIM);
    k_gemm_bf16_wmma<__hip_bfloat16, true><<<g1, 256, 0, stream>>>(
        h1, w2b, eb2, h2, BATCH, HIDDEN_DIM, HIDDEN_DIM);
    dim3 g3(LATENT_DIM / BN, BATCH / BM);
    k_gemm_bf16_wmma<float, false><<<g3, 256, 0, stream>>>(
        h2, w3b, eb3, zf, BATCH, LATENT_DIM, HIDDEN_DIM);
  }

  // Vector quantization
  k_vq<<<BATCH / VQ_ROWS, 256, 0, stream>>>(zf, emb, embT, zq_out, zqb, rowmin);

  // Decoder
  {
    dim3 g1(HIDDEN_DIM / BN, BATCH / BM);
    k_gemm_bf16_wmma<__hip_bfloat16, true><<<g1, 256, 0, stream>>>(
        zqb, d1b, db1, h1, BATCH, HIDDEN_DIM, LATENT_DIM);
    k_gemm_bf16_wmma<__hip_bfloat16, true><<<g1, 256, 0, stream>>>(
        h1, d2b, db2, h2, BATCH, HIDDEN_DIM, HIDDEN_DIM);
    dim3 g6(INPUT_DIM / BN, BATCH / BM);
    k_gemm_bf16_wmma<float, false><<<g6, 256, 0, stream>>>(
        h2, d3b, db3, x_recon, BATCH, INPUT_DIM, HIDDEN_DIM);
  }

  // Losses
  k_sqdiff_partial<<<NPART, 256, 0, stream>>>(x_recon, x, part, BATCH * INPUT_DIM);
  k_finalize<<<1, 256, 0, stream>>>(rowmin, BATCH, part, NPART, total_l, vq_l,
                                    1.0f / (float)((size_t)BATCH * INPUT_DIM),
                                    1.25f / (float)((size_t)BATCH * LATENT_DIM));
}
